// DecoderBlock_52896817217548
// MI455X (gfx1250) — matrix operand order repaired and checked by reading, not compiled, not run on hardware
//
#include <hip/hip_runtime.h>
#include <hip/hip_bf16.h>
#include <math.h>
#include <cstdint>

#define BATCH   2
#define SEQ     2048
#define DMODEL  1024
#define NHEADS  16
#define DK      64
#define DFF     4096
#define MROWS   (BATCH * SEQ)   // 4096

typedef unsigned short u16;
typedef __attribute__((ext_vector_type(16))) __bf16 v16bf;
typedef __attribute__((ext_vector_type(8)))  float  v8f;

union FragU { uint4 u[2]; v16bf v; u16 s[16]; };

__device__ __forceinline__ u16 f32_to_bf16(float f) {
  unsigned int u = __float_as_uint(f);
  unsigned int r = u + 0x7FFFu + ((u >> 16) & 1u);
  return (u16)(r >> 16);
}

__device__ __forceinline__ v8f wmma_bf16(v16bf a, v16bf b, v8f c) {
  return __builtin_amdgcn_wmma_f32_16x16x32_bf16(
      /*neg_a=*/false, a, /*neg_b=*/false, b,
      /*c_mod=*/(short)0, c, /*reuse_a=*/false, /*reuse_b=*/false);
}

// gfx1250 async LDS copy: per-lane 16B global -> LDS, tracked by ASYNCcnt.
// vdst VGPR holds the LDS byte offset (low 32 bits of the generic shared
// address per the aperture mapping), vaddr holds the 64-bit global address.
__device__ __forceinline__ void async_copy_b128(uint32_t lds_off, const void* g) {
  asm volatile("global_load_async_to_lds_b128 %0, %1, off"
               :: "v"(lds_off), "v"((unsigned long long)(uintptr_t)g)
               : "memory");
}

// Wait until ASYNCcnt <= N (async ops complete in order per wave).
template <int N>
__device__ __forceinline__ void wait_async_le() {
#if defined(__has_builtin)
#if __has_builtin(__builtin_amdgcn_s_wait_asynccnt)
  __builtin_amdgcn_s_wait_asynccnt((short)N);
  return;
#endif
#endif
  asm volatile("s_wait_asynccnt %0" :: "i"(N) : "memory");
}

// ------------------------------------------------------- convert + transpose
// in: [K,N] f32 -> out: [N,K] bf16 (LDS-tiled 32x32 transpose)
__global__ __launch_bounds__(256)
void cvt_t_kernel(const float* __restrict__ in, u16* __restrict__ out,
                  int K, int N) {
  __shared__ u16 tile[32][33];
  int bk = blockIdx.y * 32, bn = blockIdx.x * 32;
  int tx = threadIdx.x & 31, ty = threadIdx.x >> 5;   // 32 x 8
#pragma unroll
  for (int r = 0; r < 32; r += 8)
    tile[ty + r][tx] = f32_to_bf16(in[(size_t)(bk + ty + r) * N + bn + tx]);
  __syncthreads();
#pragma unroll
  for (int r = 0; r < 32; r += 8)
    out[(size_t)(bn + ty + r) * K + bk + tx] = tile[tx][ty + r];
}

// ---------------------------------------------------------------- layernorm
__global__ __launch_bounds__(256)
void ln_kernel(const float* __restrict__ x, const float* __restrict__ g,
               const float* __restrict__ bta, u16* __restrict__ out) {
  __shared__ float ssum[256];
  __shared__ float ssq[256];
  int row = blockIdx.x;
  int tid = threadIdx.x;
  const float* xr = x + (size_t)row * DMODEL;
  float vals[4];
  float s = 0.f, sq = 0.f;
#pragma unroll
  for (int k = 0; k < 4; k++) {
    float v = xr[tid + k * 256];
    vals[k] = v; s += v; sq += v * v;
  }
  ssum[tid] = s; ssq[tid] = sq;
  __syncthreads();
  for (int off = 128; off > 0; off >>= 1) {
    if (tid < off) { ssum[tid] += ssum[tid + off]; ssq[tid] += ssq[tid + off]; }
    __syncthreads();
  }
  float mu  = ssum[0] * (1.0f / DMODEL);
  float var = ssq[0] * (1.0f / DMODEL) - mu * mu;
  float rs  = rsqrtf(var + 1e-5f);
#pragma unroll
  for (int k = 0; k < 4; k++) {
    int c = tid + k * 256;
    out[(size_t)row * DMODEL + c] = f32_to_bf16((vals[k] - mu) * rs * g[c] + bta[c]);
  }
}

// ---------------------------------------------------------------- WMMA GEMM
// C[M,N] = A[M,K](bf16) @ Bt[N,K](bf16)^T + bias, per-template epilogue.
// Double-buffered: async b128 copies of tile i+1 overlap WMMAs on tile i.
// EPI 0: bf16 scatter to [B,H,S,dk]   (Q, K projections)
// EPI 1: bf16 scatter to [B,H,dk,S]   (V^T projection)
// EPI 2: f32 out = acc + bias + res   (residual adds)
// EPI 3: bf16 out = gelu(acc + bias)  (FFN hidden)
template <int EPI>
__global__ __launch_bounds__(256)
void gemm_kernel(const u16* __restrict__ A, const u16* __restrict__ Bt,
                 const float* __restrict__ bias, const float* __restrict__ res,
                 float* __restrict__ outF, u16* __restrict__ outB,
                 int M, int N, int K) {
  __shared__ __align__(16) u16 As[2][128 * 32];   // [m][k] ping-pong
  __shared__ __align__(16) u16 Bs[2][128 * 32];   // [n][k] ping-pong

  int tid  = threadIdx.x;
  int lane = tid & 31, wave = tid >> 5;
  int wr = wave >> 2, wc = wave & 3;              // 2x4 wave grid
  int lh = lane & 15, kh = lane >> 4;
  int m0 = blockIdx.y * 128, n0 = blockIdx.x * 128;

  v8f acc[4][2] = {};

  int row = tid >> 1, seg = tid & 1;              // 128 rows x (2 x 16) k
  uint32_t ldsA[2], ldsB[2];
#pragma unroll
  for (int b = 0; b < 2; b++) {
    ldsA[b] = (uint32_t)(uintptr_t)&As[b][row * 32 + seg * 16];
    ldsB[b] = (uint32_t)(uintptr_t)&Bs[b][row * 32 + seg * 16];
  }
  const u16* gaBase = A  + (size_t)(m0 + row) * K + seg * 16;
  const u16* gbBase = Bt + (size_t)(n0 + row) * K + seg * 16;

  auto issue = [&](int buf, int k0) {
    async_copy_b128(ldsA[buf],      gaBase + k0);
    async_copy_b128(ldsA[buf] + 16, gaBase + k0 + 8);
    async_copy_b128(ldsB[buf],      gbBase + k0);
    async_copy_b128(ldsB[buf] + 16, gbBase + k0 + 8);
  };

  int nk = K >> 5;                                // K / 32 tiles
  issue(0, 0);
  for (int it = 0; it < nk; it++) {
    int cur = it & 1;
    if (it + 1 < nk) {
      issue(cur ^ 1, (it + 1) * 32);              // overlap next tile's DMA
      if (it + 2 < nk) __builtin_prefetch(gaBase + (it + 2) * 32, 0, 1);
      wait_async_le<4>();                         // tile `it` landed (in-order)
    } else {
      wait_async_le<0>();
    }
    __syncthreads();                              // all waves: tile visible
    const u16* AsC = As[cur];
    const u16* BsC = Bs[cur];
#pragma unroll
    for (int mi = 0; mi < 4; mi++) {
      FragU fa;
      const u16* pa = &AsC[(wr * 64 + mi * 16 + lh) * 32 + kh * 8];
      fa.u[0] = *(const uint4*)pa;                // K = kh*8 .. +7
      fa.u[1] = *(const uint4*)(pa + 16);         // K = 16 + kh*8 .. +7
#pragma unroll
      for (int ni = 0; ni < 2; ni++) {
        FragU fb;
        const u16* pb = &BsC[(wc * 32 + ni * 16 + lh) * 32 + kh * 8];
        fb.u[0] = *(const uint4*)pb;              // K = kh*8 .. +7
        fb.u[1] = *(const uint4*)(pb + 16);       // K = 16 + kh*8 .. +7
        acc[mi][ni] = wmma_bf16(fa.v, fb.v, acc[mi][ni]);
      }
    }
    __syncthreads();                              // buffer free for reuse
  }

  // ---- epilogue
#pragma unroll
  for (int mi = 0; mi < 4; mi++) {
#pragma unroll
    for (int ni = 0; ni < 2; ni++) {
      int n = n0 + wc * 32 + ni * 16 + lh;
      float bv = bias[n];
#pragma unroll
      for (int i = 0; i < 8; i++) {
        int m = m0 + wr * 64 + mi * 16 + i + 8 * kh;
        float v = acc[mi][ni][i] + bv;
        if constexpr (EPI == 0) {                 // Q/K -> [B,H,S,dk]
          int bI = m >> 11, s = m & (SEQ - 1);
          int head = n >> 6, d = n & (DK - 1);
          outB[(((size_t)(bI * NHEADS + head)) * SEQ + s) * DK + d] = f32_to_bf16(v);
        } else if constexpr (EPI == 1) {          // V^T -> [B,H,dk,S]
          int bI = m >> 11, s = m & (SEQ - 1);
          int head = n >> 6, d = n & (DK - 1);
          outB[(((size_t)(bI * NHEADS + head)) * DK + d) * SEQ + s] = f32_to_bf16(v);
        } else if constexpr (EPI == 2) {          // f32 + residual
          size_t o = (size_t)m * N + n;
          outF[o] = v + res[o];
        } else {                                  // exact GELU -> bf16
          float gl = 0.5f * v * (1.0f + erff(v * 0.70710678118654752f));
          outB[(size_t)m * N + n] = f32_to_bf16(gl);
        }
      }
    }
  }
}

// ---------------------------------------------------------------- flash attention
// One wave per 16-query tile of one (batch, head). Causal + pad mask.
__global__ __launch_bounds__(32)
void attn_kernel(const u16* __restrict__ q, const u16* __restrict__ k,
                 const u16* __restrict__ vt, const int* __restrict__ pad,
                 u16* __restrict__ ctx) {
  __shared__ __align__(16) u16 Ps[16 * 32];

  int lane = threadIdx.x;
  int lh = lane & 15, kh = lane >> 4;
  int qb = blockIdx.x % (SEQ / 16);
  int bh = blockIdx.x / (SEQ / 16);
  int bI = bh / NHEADS, hI = bh % NHEADS;
  int q0 = qb * 16;

  const u16* qbase = q  + (size_t)bh * SEQ * DK;
  const u16* kbase = k  + (size_t)bh * SEQ * DK;
  const u16* vbase = vt + (size_t)bh * DK * SEQ;

  // Q A-fragments (two K=32 chunks of dk=64), contiguous 16B loads
  FragU qf[2];
#pragma unroll
  for (int kc = 0; kc < 2; kc++) {
    const u16* p = qbase + (size_t)(q0 + lh) * DK + kc * 32 + kh * 8;
    qf[kc].u[0] = *(const uint4*)p;
    qf[kc].u[1] = *(const uint4*)(p + 16);
  }

  float m_[8], l_[8];
  v8f O[4] = {};
#pragma unroll
  for (int i = 0; i < 8; i++) { m_[i] = -INFINITY; l_[i] = 0.f; }

  int nj = ((q0 + 15) >> 5) + 1;              // causal: kv blocks of 32
  for (int j = 0; j < nj; j++) {
    int kb = j * 32;
    // ---- scores S = Q @ K^T for 16x32 block (2 col-tiles x 2 K-chunks)
    v8f sc[2] = {};
#pragma unroll
    for (int t = 0; t < 2; t++) {
#pragma unroll
      for (int kc = 0; kc < 2; kc++) {
        FragU kf;
        const u16* p = kbase + (size_t)(kb + t * 16 + lh) * DK + kc * 32 + kh * 8;
        kf.u[0] = *(const uint4*)p;
        kf.u[1] = *(const uint4*)(p + 16);
        sc[t] = wmma_bf16(qf[kc].v, kf.v, sc[t]);
      }
    }
    int pm0 = pad[bI * SEQ + kb + lh];
    int pm1 = pad[bI * SEQ + kb + 16 + lh];
    // ---- online softmax (row = i + 8*kh across vgprs; col = lane%16)
    float pr0[8], pr1[8];
#pragma unroll
    for (int i = 0; i < 8; i++) {
      int row = q0 + i + 8 * kh;
      int c0 = kb + lh, c1 = kb + 16 + lh;
      float v0 = sc[0][i] * 0.125f;           // 1/sqrt(64)
      float v1 = sc[1][i] * 0.125f;
      if (c0 > row || pm0 == 0) v0 = -INFINITY;
      if (c1 > row || pm1 == 0) v1 = -INFINITY;
      float bm = fmaxf(v0, v1);
      bm = fmaxf(bm, __shfl_xor(bm, 1));
      bm = fmaxf(bm, __shfl_xor(bm, 2));
      bm = fmaxf(bm, __shfl_xor(bm, 4));
      bm = fmaxf(bm, __shfl_xor(bm, 8));
      float nm = fmaxf(m_[i], bm);
      float alpha = __expf(m_[i] - nm);
      float p0 = __expf(v0 - nm);
      float p1 = __expf(v1 - nm);
      float rs = p0 + p1;
      rs += __shfl_xor(rs, 1);
      rs += __shfl_xor(rs, 2);
      rs += __shfl_xor(rs, 4);
      rs += __shfl_xor(rs, 8);
      l_[i] = l_[i] * alpha + rs;
      m_[i] = nm;
#pragma unroll
      for (int nc = 0; nc < 4; nc++) O[nc][i] *= alpha;
      pr0[i] = p0; pr1[i] = p1;
    }
    // ---- C-layout -> A-layout for P via LDS (bf16)
#pragma unroll
    for (int i = 0; i < 8; i++) {
      int r = i + 8 * kh;
      Ps[r * 32 + lh]      = f32_to_bf16(pr0[i]);
      Ps[r * 32 + 16 + lh] = f32_to_bf16(pr1[i]);
    }
    FragU pf;
    {
      const u16* p = &Ps[lh * 32 + kh * 8];
      pf.u[0] = *(const uint4*)p;
      pf.u[1] = *(const uint4*)(p + 16);
    }
    // ---- O += P @ V  (V^T layout -> contiguous B-fragments along kv)
#pragma unroll
    for (int nc = 0; nc < 4; nc++) {
      FragU vf;
      const u16* p = vbase + (size_t)(nc * 16 + lh) * SEQ + kb + kh * 8;
      vf.u[0] = *(const uint4*)p;
      vf.u[1] = *(const uint4*)(p + 16);
      O[nc] = wmma_bf16(pf.v, vf.v, O[nc]);
    }
  }
  // ---- finalize: ctx[b, s, h*64 + d] = O / l
#pragma unroll
  for (int nc = 0; nc < 4; nc++) {
#pragma unroll
    for (int i = 0; i < 8; i++) {
      float inv = 1.0f / l_[i];
      size_t off = ((size_t)(bI * SEQ + q0 + i + 8 * kh)) * DMODEL + hI * DK + nc * 16 + lh;
      ctx[off] = f32_to_bf16(O[nc][i] * inv);
    }
  }
}

// ---------------------------------------------------------------- launch
extern "C" void kernel_launch(void* const* d_in, const int* in_sizes, int n_in,
                              void* d_out, int out_size, void* d_ws, size_t ws_size,
                              hipStream_t stream) {
  const float* x    = (const float*)d_in[0];
  const int*   pad  = (const int*)d_in[1];
  const float* ln1g = (const float*)d_in[2];
  const float* ln1b = (const float*)d_in[3];
  const float* ln2g = (const float*)d_in[4];
  const float* ln2b = (const float*)d_in[5];
  const float* wq = (const float*)d_in[6];   const float* bq = (const float*)d_in[7];
  const float* wk = (const float*)d_in[8];   const float* bk = (const float*)d_in[9];
  const float* wv = (const float*)d_in[10];  const float* bv = (const float*)d_in[11];
  const float* wo = (const float*)d_in[12];  const float* bo = (const float*)d_in[13];
  const float* w1 = (const float*)d_in[14];  const float* b1 = (const float*)d_in[15];
  const float* w2 = (const float*)d_in[16];  const float* b2 = (const float*)d_in[17];

  char* ws = (char*)d_ws;
  size_t off = 0;
  auto alloc = [&](size_t bytes) -> void* {
    void* p = ws + off;
    off += (bytes + 255) & ~(size_t)255;
    return p;
  };
  const size_t MD = (size_t)MROWS * DMODEL;
  u16*   hB   = (u16*)alloc(MD * 2);
  u16*   wqT  = (u16*)alloc((size_t)DMODEL * DMODEL * 2);
  u16*   wkT  = (u16*)alloc((size_t)DMODEL * DMODEL * 2);
  u16*   wvT  = (u16*)alloc((size_t)DMODEL * DMODEL * 2);
  u16*   woT  = (u16*)alloc((size_t)DMODEL * DMODEL * 2);
  u16*   w1T  = (u16*)alloc((size_t)DMODEL * DFF * 2);   // [F, D]
  u16*   w2T  = (u16*)alloc((size_t)DFF * DMODEL * 2);   // [D, F]
  u16*   qB   = (u16*)alloc(MD * 2);
  u16*   kB   = (u16*)alloc(MD * 2);
  u16*   vtB  = (u16*)alloc(MD * 2);
  u16*   ctxB = (u16*)alloc(MD * 2);
  float* x2   = (float*)alloc(MD * 4);
  u16*   h2B  = (u16*)alloc(MD * 2);
  u16*   gB   = (u16*)alloc((size_t)MROWS * DFF * 2);

  // weights: fp32 [K,N] -> bf16 [N,K] (transposed so GEMM B-tiles are
  // K-contiguous and can be async-copied to LDS)
  auto cvtT = [&](const float* src, u16* dst, int K, int N) {
    dim3 g(N / 32, K / 32);
    cvt_t_kernel<<<g, 256, 0, stream>>>(src, dst, K, N);
  };
  cvtT(wq, wqT, DMODEL, DMODEL);
  cvtT(wk, wkT, DMODEL, DMODEL);
  cvtT(wv, wvT, DMODEL, DMODEL);
  cvtT(wo, woT, DMODEL, DMODEL);
  cvtT(w1, w1T, DMODEL, DFF);
  cvtT(w2, w2T, DFF, DMODEL);

  // LN1 -> h (bf16)
  ln_kernel<<<MROWS, 256, 0, stream>>>(x, ln1g, ln1b, hB);

  dim3 blk(256);
  dim3 gD(DMODEL / 128, MROWS / 128);
  dim3 gF(DFF / 128,    MROWS / 128);

  // QKV projections (V stored transposed per head)
  gemm_kernel<0><<<gD, blk, 0, stream>>>(hB, wqT, bq, nullptr, nullptr, qB,
                                         MROWS, DMODEL, DMODEL);
  gemm_kernel<0><<<gD, blk, 0, stream>>>(hB, wkT, bk, nullptr, nullptr, kB,
                                         MROWS, DMODEL, DMODEL);
  gemm_kernel<1><<<gD, blk, 0, stream>>>(hB, wvT, bv, nullptr, nullptr, vtB,
                                         MROWS, DMODEL, DMODEL);

  // flash attention
  attn_kernel<<<BATCH * NHEADS * (SEQ / 16), 32, 0, stream>>>(qB, kB, vtB, pad, ctxB);

  // output projection + residual: x2 = ctx @ wo + bo + x
  gemm_kernel<2><<<gD, blk, 0, stream>>>(ctxB, woT, bo, x, x2, nullptr,
                                         MROWS, DMODEL, DMODEL);

  // LN2 -> h2 (bf16)
  ln_kernel<<<MROWS, 256, 0, stream>>>(x2, ln2g, ln2b, h2B);

  // FFN: g = gelu(h2 @ w1 + b1); out = g @ w2 + b2 + x2
  gemm_kernel<3><<<gF, blk, 0, stream>>>(h2B, w1T, b1, nullptr, nullptr, gB,
                                         MROWS, DFF, DMODEL);
  gemm_kernel<2><<<gD, blk, 0, stream>>>(gB, w2T, b2, x2, (float*)d_out, nullptr,
                                         MROWS, DMODEL, DFF);
}
